// HashGridINR_32598801777091
// MI455X (gfx1250) — matrix operand order repaired and checked by reading, not compiled, not run on hardware
//
#include <hip/hip_runtime.h>
#include <hip/hip_bf16.h>

// ---------------------------------------------------------------------------
// Instant-NGP hash-grid encode + 33->256->256->256->256->1 MLP, fused.
// MI455X (gfx1250): wave32, v_wmma_f32_16x16x32_f16 tensor path.
// Round 4: async global->LDS staging of B columns (ASYNCcnt, 3-deep rotation),
// packed-f16 ReLU epilogue (cvt_pk_rtz + v_pk_max), A resident in VGPRs.
// ---------------------------------------------------------------------------

typedef _Float16 v16h __attribute__((ext_vector_type(16)));
typedef float    v8f  __attribute__((ext_vector_type(8)));
typedef int      v4i  __attribute__((vector_size(4 * sizeof(int))));

#define N_POINTS 262144
#define MTILE    256          // rows (points) per block
#define TPB      128          // 4 waves of 32; each wave owns 4 row-blocks
#define ACT_BYTES (16 * 8 * 1024)   // 16 mt x 8 kt x 1KB A-swizzled activations
#define STAGE_BYTES (3 * 8 * 1024)  // 3-deep rotating B column stage
#define SMEM_BYTES (ACT_BYTES + STAGE_BYTES)   // 152 KB

// swizzled weight offsets in d_ws (in _Float16 elements)
#define OFFW0 0               // 64x256   (K padded 33->64)
#define OFFW1 16384           // 256x256
#define OFFW2 81920
#define OFFW3 147456
#define OFFW4 212992          // 256x16   (N padded 1->16)

#if defined(__has_builtin)
#if __has_builtin(__builtin_amdgcn_global_load_async_to_lds_b128) && \
    __has_builtin(__builtin_amdgcn_s_wait_asynccnt)
#define HAVE_ASYNC_LDS 1
#endif
#endif
#ifndef HAVE_ASYNC_LDS
#define HAVE_ASYNC_LDS 0
#endif

#if HAVE_ASYNC_LDS
typedef __attribute__((address_space(1))) v4i as1_v4i;
typedef __attribute__((address_space(3))) v4i as3_v4i;
static __device__ __forceinline__ void async_cp16(const void* g, void* l) {
  // GLOBAL_LOAD_ASYNC_TO_LDS_B128: 16B per lane, global -> LDS, no VGPR data
  __builtin_amdgcn_global_load_async_to_lds_b128((as1_v4i*)g, (as3_v4i*)l, 0, 0);
}
#endif

static __device__ __forceinline__ v8f wmma16(v16h a, v16h b, v8f c) {
  // D = A(16x32 f16) * B(32x16 f16) + C(16x16 f32)
  return __builtin_amdgcn_wmma_f32_16x16x32_f16(
      false, a, false, b, (short)0, c, false, false);
}

// ---------------------------------------------------------------------------
// Weight prep: f32 [Ksrc x Nsrc] row-major -> f16 B-fragment-swizzled tiles.
// Tile (nt,kt) is a contiguous 1KB block: half index = lane*16 + i, with
//   k = kt*32 + (i&7) + 16*(i>=8) + 8*(lane>=16),  n = nt*16 + (lane&15)
// Tiles are kt-minor, so one B column (fixed nt) is KT contiguous KB.
// ---------------------------------------------------------------------------
__global__ void swz_weights_kernel(const float* __restrict__ W,
                                   _Float16* __restrict__ dst,
                                   int K, int Nn, int Ksrc, int Nsrc) {
  int idx = blockIdx.x * 256 + threadIdx.x;
  if (idx >= K * Nn) return;
  int b  = idx >> 9;                 // 512 halves per 32x16 tile
  int w  = idx & 511;
  int l  = w >> 4, i = w & 15;
  int KT = K >> 5;
  int nt = b / KT, kt = b - nt * KT;
  int k  = kt * 32 + (i & 7) + ((i >> 3) << 4) + ((l >> 4) << 3);
  int n  = nt * 16 + (l & 15);
  float v = (k < Ksrc && n < Nsrc) ? W[k * Nsrc + n] : 0.0f;
  dst[idx] = (_Float16)v;
}

// ---------------------------------------------------------------------------
// One dense layer: A (LDS A-swizzle, KT k-tiles) x W[32KT x 256] + b, ReLU,
// scatter back into LDS in next-layer A swizzle (KT=8 layout).
// A for the wave's 4 row-blocks lives entirely in VGPRs (loaded once).
// B columns staged into LDS for the whole block, async (3-deep) or via
// registers (2-deep fallback), always prefetched ahead of use.
// ---------------------------------------------------------------------------
template <int KT>
static __device__ __forceinline__ void dense_layer(char* smem,
                                                   const _Float16* __restrict__ Wz,
                                                   const float* __restrict__ bias,
                                                   int wave, int lane) {
  constexpr int CH = (KT * 1024) / (TPB * 16);   // 16B chunks per thread per column
  const int tid = wave * 32 + lane;
  char* stage0 = smem + ACT_BYTES;

#if HAVE_ASYNC_LDS
  // async-issue column 0 -> stage buffer 0
#pragma unroll
  for (int c = 0; c < CH; ++c)
    async_cp16((const char*)Wz + (tid + c * TPB) * 16,
               stage0 + (tid + c * TPB) * 16);
#else
  uint4 sreg[CH];
  {
    const uint4* src = (const uint4*)Wz;
#pragma unroll
    for (int c = 0; c < CH; ++c) sreg[c] = src[tid + c * TPB];
  }
#endif

  // A fragments -> registers (4 row-blocks x KT k-tiles)
  v16h a[4][KT];
#pragma unroll
  for (int m = 0; m < 4; ++m) {
    const int mt = wave * 4 + m;
#pragma unroll
    for (int kt = 0; kt < KT; ++kt)
      a[m][kt] = *(const v16h*)(smem + ((mt * KT + kt) << 10) + lane * 32);
  }
  __syncthreads();   // all waves hold their A in regs; LDS act is writable now

  const int nlo = lane & 15;
  const int hs  = lane >> 4;
  int cur = 0;

  for (int nt = 0; nt < 16; ++nt) {
#if HAVE_ASYNC_LDS
    char* buf = stage0 + (cur << 13);
    const int nxt = (cur == 2) ? 0 : cur + 1;
    if (nt < 15) {
      // async-issue column nt+1 into the third buffer; its last readers were
      // at iteration nt-2, fenced by barrier nt-1 -> no WAR hazard.
      char* nbuf = stage0 + (nxt << 13);
      const char* src = (const char*)(Wz + (nt + 1) * (KT * 512));
#pragma unroll
      for (int c = 0; c < CH; ++c)
        async_cp16(src + (tid + c * TPB) * 16, nbuf + (tid + c * TPB) * 16);
      __builtin_amdgcn_s_wait_asynccnt(CH);  // column nt landed; nt+1 in flight
    } else {
      __builtin_amdgcn_s_wait_asynccnt(0);
    }
#else
    char* buf = stage0 + ((nt & 1) << 13);
    {
      uint4* dst = (uint4*)buf;           // commit prefetched column nt
#pragma unroll
      for (int c = 0; c < CH; ++c) dst[tid + c * TPB] = sreg[c];
    }
    if (nt < 15) {                        // prefetch column nt+1
      const uint4* src = (const uint4*)(Wz + (nt + 1) * (KT * 512));
#pragma unroll
      for (int c = 0; c < CH; ++c) sreg[c] = src[tid + c * TPB];
    }
#endif
    __syncthreads();   // stage[buf] visible to all waves

    const int   n  = nt * 16 + nlo;
    const float bv = bias[n];
    v8f binit;
#pragma unroll
    for (int q = 0; q < 8; ++q) binit[q] = bv;     // bias folded into C
    v8f acc[4];
#pragma unroll
    for (int m = 0; m < 4; ++m) acc[m] = binit;

#pragma unroll
    for (int kt = 0; kt < KT; ++kt) {
      v16h bf = *(const v16h*)(buf + (kt << 10) + lane * 32);
#pragma unroll
      for (int m = 0; m < 4; ++m) acc[m] = wmma16(a[m][kt], bf, acc[m]);
    }

    // packed ReLU + f16 + scatter into next layer's A-fragment swizzle (KT=8)
    const int krel = n & 31;
    const int ktn  = n >> 5;
    const int h8   = (krel >> 3) & 1;
    const int bse  = krel - (h8 << 3);
    const int ii   = (bse < 8) ? bse : bse - 8;
#pragma unroll
    for (int m = 0; m < 4; ++m) {
      const int mt = wave * 4 + m;
      char* dstb = smem + ((mt * 8 + ktn) << 10) + ii * 2;
#pragma unroll
      for (int v = 0; v < 8; v += 2) {
        auto pk = __builtin_amdgcn_cvt_pkrtz(acc[m][v], acc[m][v + 1]);
        decltype(pk) hz = {};
#if defined(__has_builtin) && __has_builtin(__builtin_elementwise_maxnum)
        pk = __builtin_elementwise_maxnum(pk, hz);   // v_pk_max f16 (ReLU)
#else
        pk = __builtin_elementwise_max(pk, hz);      // v_pk_max f16 (ReLU)
#endif
        const int dl = v + hs * 8 + (h8 << 4);
        *(_Float16*)(dstb + dl * 32)       = (_Float16)pk[0];
        *(_Float16*)(dstb + (dl + 1) * 32) = (_Float16)pk[1];
      }
    }
#if HAVE_ASYNC_LDS
    cur = nxt;
#endif
  }
  __syncthreads();   // outputs visible before next layer's A loads
}

static __device__ __forceinline__ void final_layer(char* smem,
                                                   const _Float16* __restrict__ Wz,
                                                   const float* __restrict__ b4,
                                                   float* __restrict__ out,
                                                   int rowbase, int wave, int lane) {
  const int tid = wave * 32 + lane;
  char* stage0 = smem + ACT_BYTES;
  // stage the single 8KB B column (N padded 1->16)
  {
    const uint4* src = (const uint4*)Wz;
    uint4* dst = (uint4*)stage0;
#pragma unroll
    for (int c = 0; c < 4; ++c) dst[tid + c * TPB] = src[tid + c * TPB];
  }
  v16h a[4][8];
#pragma unroll
  for (int m = 0; m < 4; ++m) {
    const int mt = wave * 4 + m;
#pragma unroll
    for (int kt = 0; kt < 8; ++kt)
      a[m][kt] = *(const v16h*)(smem + ((mt * 8 + kt) << 10) + lane * 32);
  }
  __syncthreads();

  v8f acc[4];
  const v8f vz = {};
#pragma unroll
  for (int m = 0; m < 4; ++m) acc[m] = vz;
#pragma unroll
  for (int kt = 0; kt < 8; ++kt) {
    v16h bf = *(const v16h*)(stage0 + (kt << 10) + lane * 32);
#pragma unroll
    for (int m = 0; m < 4; ++m) acc[m] = wmma16(a[m][kt], bf, acc[m]);
  }
  const float bv = b4[0];
  const int hs = lane >> 4;
  if ((lane & 15) == 0) {            // column 0 holds the real output
#pragma unroll
    for (int m = 0; m < 4; ++m)
#pragma unroll
      for (int v = 0; v < 8; ++v)
        out[rowbase + (wave * 4 + m) * 16 + v + hs * 8] = acc[m][v] + bv;
  }
}

// ---------------------------------------------------------------------------
// Fused kernel: hash encode (2 points/thread) -> swizzled LDS -> 5 GEMM layers
// ---------------------------------------------------------------------------
extern "C" __global__ __launch_bounds__(TPB, 1)
void hashgrid_mlp_kernel(const float* __restrict__ x,
                         const float* __restrict__ tables,
                         const _Float16* __restrict__ Wz,
                         const float* __restrict__ b0, const float* __restrict__ b1,
                         const float* __restrict__ b2, const float* __restrict__ b3,
                         const float* __restrict__ b4,
                         float* __restrict__ out) {
  __shared__ __align__(128) char smem[SMEM_BYTES];
  const int tid  = threadIdx.x;
  const int lane = tid & 31;
  const int wave = tid >> 5;

  // Zero the kt=1 tiles of the layer-0 layout (pad features 33..63)
  {
    uint4 z4 = {0u, 0u, 0u, 0u};
    for (int j = tid; j < 16 * 64; j += TPB) {
      int mtb = j >> 6, q = j & 63;
      ((uint4*)(smem + ((mtb * 2 + 1) << 10)))[q] = z4;
    }
  }
  __syncthreads();

  // ---- hash-grid encode: 2 points per thread ----
  static constexpr float    RESF[16] = {8, 10, 13, 18, 24, 32, 42, 55,
                                        73, 97, 128, 168, 222, 294, 388, 512};
  static constexpr unsigned HS[16]   = {64, 100, 169, 324, 576, 1024, 1764, 3025,
                                        4096, 4096, 4096, 4096, 4096, 4096, 4096, 4096};
#pragma unroll
  for (int pp = 0; pp < 2; ++pp) {
    const int   r  = tid + pp * TPB;            // local row 0..255
    const int   p  = blockIdx.x * MTILE + r;
    const float c0 = x[p * 3 + 0];
    const float c1 = x[p * 3 + 1];
    const float c2 = x[p * 3 + 2];
    const int   rl = r & 15;
    const int   mt = r >> 4;
    char* ab0 = smem + ((mt * 2 + 0) << 10);    // kt=0 tile: features 0..31
#pragma unroll
    for (int l = 0; l < 16; ++l) {
      const float    rr  = RESF[l];
      const unsigned hsz = HS[l];
      float sx = c0 * rr, sy = c1 * rr;
      float fx = floorf(sx), fy = floorf(sy);
      float tx = sx - fx,   ty = sy - fy;
      int   ix = (int)fx,   iy = (int)fy;
      float f0 = 0.0f, f1 = 0.0f;
#pragma unroll
      for (int c = 0; c < 4; ++c) {
        const int ca = c >> 1, cb = c & 1;
        unsigned u0 = (unsigned)(ix + ca);
        unsigned u1 = (unsigned)(iy + cb) * 2654435761u;
        unsigned h  = (u0 ^ u1) % hsz;           // folds to AND / magic-mul
        const float2 t2 = *(const float2*)(tables + (l * 4096u + h) * 2u);
        float w = (ca ? tx : 1.0f - tx) * (cb ? ty : 1.0f - ty);
        f0 = fmaf(w, t2.x, f0);
        f1 = fmaf(w, t2.y, f1);
      }
      // pack the level's feature pair and scatter into A-fragment swizzle
      const int krel = 2 * l;
      const int h8   = (krel >> 3) & 1;
      const int bse  = krel - (h8 << 3);
      const int i0   = (bse < 8) ? bse : bse - 8;  // even -> dword aligned
      const int dl   = rl + (h8 << 4);
      union { _Float16 f[2]; unsigned u; } pk;
      pk.f[0] = (_Float16)f0;
      pk.f[1] = (_Float16)f1;
      *(unsigned*)(ab0 + dl * 32 + i0 * 2) = pk.u;
    }
    // feature 32 = x[:,2] -> kt=1 tile, krel=0 (h8=0, i=0)
    *(_Float16*)(smem + ((mt * 2 + 1) << 10) + rl * 32) = (_Float16)c2;
  }
  __syncthreads();

  // ---- MLP ----
  dense_layer<2>(smem, Wz + OFFW0, b0, wave, lane);  // 64(=33 pad) -> 256
  dense_layer<8>(smem, Wz + OFFW1, b1, wave, lane);  // 256 -> 256
  dense_layer<8>(smem, Wz + OFFW2, b2, wave, lane);  // 256 -> 256
  dense_layer<8>(smem, Wz + OFFW3, b3, wave, lane);  // 256 -> 256
  final_layer(smem, Wz + OFFW4, b4, out, blockIdx.x * MTILE, wave, lane);
}

// ---------------------------------------------------------------------------
extern "C" void kernel_launch(void* const* d_in, const int* in_sizes, int n_in,
                              void* d_out, int out_size, void* d_ws, size_t ws_size,
                              hipStream_t stream) {
  const float* x      = (const float*)d_in[0];
  const float* tables = (const float*)d_in[1];
  const float* W0 = (const float*)d_in[2];
  const float* b0 = (const float*)d_in[3];
  const float* W1 = (const float*)d_in[4];
  const float* b1 = (const float*)d_in[5];
  const float* W2 = (const float*)d_in[6];
  const float* b2 = (const float*)d_in[7];
  const float* W3 = (const float*)d_in[8];
  const float* b3 = (const float*)d_in[9];
  const float* W4 = (const float*)d_in[10];
  const float* b4 = (const float*)d_in[11];
  float*    out = (float*)d_out;
  _Float16* Wz  = (_Float16*)d_ws;   // ~424 KB of swizzled f16 weights

  swz_weights_kernel<<<(64 * 256) / 256, 256, 0, stream>>>(W0, Wz + OFFW0, 64, 256, 33, 256);
  swz_weights_kernel<<<(256 * 256) / 256, 256, 0, stream>>>(W1, Wz + OFFW1, 256, 256, 256, 256);
  swz_weights_kernel<<<(256 * 256) / 256, 256, 0, stream>>>(W2, Wz + OFFW2, 256, 256, 256, 256);
  swz_weights_kernel<<<(256 * 256) / 256, 256, 0, stream>>>(W3, Wz + OFFW3, 256, 256, 256, 256);
  swz_weights_kernel<<<(256 * 16) / 256, 256, 0, stream>>>(W4, Wz + OFFW4, 256, 16, 256, 1);

  hashgrid_mlp_kernel<<<N_POINTS / MTILE, TPB, 0, stream>>>(
      x, tables, Wz, b0, b1, b2, b3, b4, out);
}
